// ConvBlockE3_78365973283119
// MI455X (gfx1250) — matrix operand order repaired and checked by reading, not compiled, not run on hardware
//
#include <hip/hip_runtime.h>
#include <math.h>

#define EDGES 160000
#define NODES 10000
#define CCH   128
#define DFULL 512

static constexpr float F_INV_SQRT3 = 0.5773502691896258f;
static constexpr float F_INV_SQRT2 = 0.7071067811865476f;
static constexpr float F_LIN_SCALE = 0.08838834764831845f; // 1/sqrt(128)

// Swizzled-fragment storage: each lane's 16 bf16 operand values are contiguous,
// with a per-lane stride of 20 elements (40B = 10 dwords -> conflict-free b64s).
#define FSTRIDE 20
#define SROW    516   // f32 row stride for 512-col node tiles (bank padding)

typedef __attribute__((ext_vector_type(16))) __bf16 v16bf;
typedef __attribute__((ext_vector_type(8)))  float  v8f;
typedef __attribute__((ext_vector_type(2)))  float  v2f;

union BFrag {
    v16bf v;
    __bf16 b[16];
    unsigned long long q[4];
};

__device__ inline v8f wmma_bf16(v16bf a, v16bf b, v8f c) {
    return __builtin_amdgcn_wmma_f32_16x16x32_bf16(false, a, false, b, (short)0, c,
                                                   false, false);
}

// Load one pre-swizzled fragment: 4x ds_load_b64, no packing movs.
__device__ inline v16bf load_frag(const __bf16* base, int fragIdx, int lane) {
    const unsigned long long* p =
        (const unsigned long long*)(base + (size_t)(fragIdx * 32 + lane) * FSTRIDE);
    BFrag f;
    f.q[0] = p[0]; f.q[1] = p[1]; f.q[2] = p[2]; f.q[3] = p[3];
    return f.v;
}

// B-matrix (32x16 bf16) swizzle: element (k in 0..31, n in 0..15) of fragment
// (kb,nb) lives at lane = ((k>>3)&1)*16 + (n&15), slot = (k&7) + 8*(k>>4).
__device__ inline int b_slot_lane(int kk, int n) { return (((kk >> 3) & 1) << 4) + (n & 15); }

// A-matrix (16x32 bf16) swizzle: element (row, k in 0..31) lives at
// lane = row + 16*((k>>3)&1), slot = 2*(((k>>4)&1)*4 + ((k>>1)&3)) + (k&1).
__device__ inline int a_slot(int kk)           { return (((kk >> 4) & 1) << 3) + ((kk >> 1) & 3) * 2 + (kk & 1); }
__device__ inline int a_lane(int row, int kk)  { return row + (((kk >> 3) & 1) << 4); }

// First-layer A from edge_embedding (E x 8), K zero-padded to 32 (registers).
__device__ inline v16bf load_a_emb(const float* __restrict__ emb, int e0, int m, int hi) {
    BFrag f;
#pragma unroll
    for (int p = 0; p < 8; ++p) {
        int k = ((p < 4) ? 0 : 16) + hi * 8 + (p & 3) * 2;
        float v0 = (k     < 8) ? emb[(e0 + m) * 8 + k]     : 0.f;
        float v1 = (k + 1 < 8) ? emb[(e0 + m) * 8 + k + 1] : 0.f;
        f.b[2 * p]     = (__bf16)v0;
        f.b[2 * p + 1] = (__bf16)v1;
    }
    return f.v;
}

// Store a D tile (silu applied) directly into A-swizzled bf16 activation scratch.
__device__ inline void store_silu_a(__bf16* h, v8f c, int n0, int m, int hi) {
    const int col  = n0 + m;               // K index of this value in next layer's A
    const int kb   = col >> 5;
    const int kk   = col & 31;
    const int slot = a_slot(kk);
    const int hiA  = ((kk >> 3) & 1) << 4;
#pragma unroll
    for (int r = 0; r < 8; ++r) {
        float x = c[r];
        float s = x * __builtin_amdgcn_rcpf(1.f + __expf(-x));
        int lane_a = (r + 8 * hi) + hiA;   // row + 16*hiA
        h[(kb * 32 + lane_a) * FSTRIDE + slot] = (__bf16)s;
    }
}

// ---------------------------------------------------------------------------
__global__ void zero_kernel(float* __restrict__ p, int n) {
    int i = blockIdx.x * blockDim.x + threadIdx.x;
    if (i < n) p[i] = 0.f;
}

// ---------------------------------------------------------------------------
// Edge kernel: radial MLP (WMMA bf16) + message formation + atomic scatter-add
// 8 waves/WG, each wave owns one 16-edge tile.
// ---------------------------------------------------------------------------
__global__ __launch_bounds__(256) void edge_kernel(
    const float* __restrict__ nf,   const float* __restrict__ ef,
    const float* __restrict__ attrs,const float* __restrict__ emb,
    const int*   __restrict__ snd,  const int*   __restrict__ rcv,
    const float* __restrict__ Wm1,  const float* __restrict__ Wm2,
    const float* __restrict__ Wm3,  const float* __restrict__ Wm4,
    float* __restrict__ agg)
{
    // Fragment counts: W1: 1kb x 4nb; W2/W3: 2kb x 4nb; W4: 2kb x 40nb.
    __shared__ __attribute__((aligned(16))) __bf16 sW1[ 4 * 32 * FSTRIDE];
    __shared__ __attribute__((aligned(16))) __bf16 sW2[ 8 * 32 * FSTRIDE];
    __shared__ __attribute__((aligned(16))) __bf16 sW3[ 8 * 32 * FSTRIDE];
    __shared__ __attribute__((aligned(16))) __bf16 sW4[80 * 32 * FSTRIDE];
    __shared__ __attribute__((aligned(16))) __bf16 sH[8][2 * 32 * FSTRIDE]; // per-wave A scratch

    const int tid = threadIdx.x;
    // --- stage weights into swizzled-fragment LDS layout ---
    for (int i = tid; i < 32 * 64; i += 256) {           // Wm1 (8x64, K pad 32)
        int k = i >> 6, n = i & 63;
        float w = (k < 8) ? Wm1[k * 64 + n] : 0.f;
        sW1[((n >> 4) * 32 + b_slot_lane(k, n)) * FSTRIDE + ((k & 7) + ((k >> 4) << 3))] = (__bf16)w;
    }
    for (int i = tid; i < 64 * 64; i += 256) {           // Wm2 / Wm3 (64x64)
        int k = i >> 6, n = i & 63;
        int kb = k >> 5, kk = k & 31;
        int pos = ((kb * 4 + (n >> 4)) * 32 + b_slot_lane(kk, n)) * FSTRIDE + ((kk & 7) + ((kk >> 4) << 3));
        sW2[pos] = (__bf16)Wm2[i];
        sW3[pos] = (__bf16)Wm3[i];
    }
    for (int i = tid; i < 64 * 640; i += 256) {          // Wm4 (64x640)
        int k = i / 640, n = i - k * 640;
        int kb = k >> 5, kk = k & 31;
        int pos = ((kb * 40 + (n >> 4)) * 32 + b_slot_lane(kk, n)) * FSTRIDE + ((kk & 7) + ((kk >> 4) << 3));
        sW4[pos] = (__bf16)Wm4[i];
    }

    const int wave = tid >> 5, lane = tid & 31;
    const int m = lane & 15, hi = lane >> 4;
    const int tile = blockIdx.x * 8 + wave;
    const int e0 = tile * 16;

    // Prefetch this wave's edge_features tile (16 rows x 512 f32 = 32 KB):
    // 256 cachelines, 8 per lane, overlapped with the MLP.
    if (e0 < EDGES) {
        const float* efb = ef + (size_t)e0 * DFULL;
#pragma unroll
        for (int q = 0; q < 8; ++q)
            __builtin_prefetch(efb + (size_t)(q * 32 + lane) * 32, 0, 0);
    }
    __syncthreads();
    if (e0 >= EDGES) return;
    __bf16* h = sH[wave];

    // ---- layer 1: (16x8 pad 32) @ (8x64) -> silu -> h (A-swizzled) ----
    {
        v16bf a = load_a_emb(emb, e0, m, hi);
#pragma unroll
        for (int nt = 0; nt < 4; ++nt) {
            v8f c = {};
            c = wmma_bf16(a, load_frag(sW1, nt, lane), c);
            store_silu_a(h, c, nt * 16, m, hi);
        }
    }
    // ---- layers 2 & 3: (16x64) @ (64x64) -> silu ----
    {
        const __bf16* Ws[2] = { sW2, sW3 };
#pragma unroll
        for (int L = 0; L < 2; ++L) {
            v16bf a0 = load_frag(h, 0, lane);
            v16bf a1 = load_frag(h, 1, lane);
#pragma unroll
            for (int nt = 0; nt < 4; ++nt) {
                v8f c = {};
                c = wmma_bf16(a0, load_frag(Ws[L], 0 * 4 + nt, lane), c);
                c = wmma_bf16(a1, load_frag(Ws[L], 1 * 4 + nt, lane), c);
                store_silu_a(h, c, nt * 16, m, hi);
            }
        }
    }
    // ---- layer 4 + message formation, 16-channel blocks ----
    v16bf a0 = load_frag(h, 0, lane);
    v16bf a1 = load_frag(h, 1, lane);

    // Hoist per-edge metadata for this lane's 8 edge rows out of the cb loop.
    int   seA[8], reA[8];
    float s0A[8], sxA[8], syA[8], szA[8];
#pragma unroll
    for (int r = 0; r < 8; ++r) {
        const int e = e0 + r + 8 * hi;
        seA[r] = snd[e];
        reA[r] = rcv[e];
        s0A[r] = attrs[e * 4 + 0];
        sxA[r] = attrs[e * 4 + 1];
        syA[r] = attrs[e * 4 + 2];
        szA[r] = attrs[e * 4 + 3];
    }

    for (int cb = 0; cb < 8; ++cb) {
        const int c0 = cb * 16;
        v8f d[5];
#pragma unroll
        for (int j = 0; j < 5; ++j) {
            const int nb = j * 8 + cb;            // column block (j*128+c0)/16
            v8f acc = {};
            acc = wmma_bf16(a0, load_frag(sW4, 0 * 40 + nb, lane), acc);
            acc = wmma_bf16(a1, load_frag(sW4, 1 * 40 + nb, lane), acc);
            d[j] = acc;
        }
        // D layout: VGPR r -> edge row (r + 8*hi), lane%16 -> channel (c0 + m)
        const int c = c0 + m;
        const int bi = CCH + c * 3;
#pragma unroll
        for (int r = 0; r < 8; ++r) {
            const int e  = e0 + r + 8 * hi;
            const float s0 = s0A[r], s1x = sxA[r], s1y = syA[r], s1z = szA[r];
            const float* ns = nf + (size_t)seA[r] * DFULL;
            const float* nr = nf + (size_t)reA[r] * DFULL;
            const float* ee = ef + (size_t)e * DFULL;
            const float x0 = ns[c] + nr[c] + ee[c];
            const float xx = ns[bi]     + nr[bi]     + ee[bi];
            const float xy = ns[bi + 1] + nr[bi + 1] + ee[bi + 1];
            const float xz = ns[bi + 2] + nr[bi + 2] + ee[bi + 2];
            const float w0 = d[0][r], w1 = d[1][r], w2 = d[2][r],
                        w3 = d[3][r], w4 = d[4][r];
            const float dot = xx * s1x + xy * s1y + xz * s1z;
            const float cx  = xy * s1z - xz * s1y;
            const float cy  = xz * s1x - xx * s1z;
            const float cz  = xx * s1y - xy * s1x;
            const float m0v = w0 * x0 * s0 + w1 * dot * F_INV_SQRT3;
            const float m1x = w2 * xx * s0 + w3 * x0 * s1x + w4 * cx * F_INV_SQRT2;
            const float m1y = w2 * xy * s0 + w3 * x0 * s1y + w4 * cy * F_INV_SQRT2;
            const float m1z = w2 * xz * s0 + w3 * x0 * s1z + w4 * cz * F_INV_SQRT2;
            float* ar = agg + (size_t)reA[r] * DFULL;
            atomicAdd(ar + c,      m0v);
            atomicAdd(ar + bi,     m1x);
            atomicAdd(ar + bi + 1, m1y);
            atomicAdd(ar + bi + 2, m1z);
        }
    }
}

// ---------------------------------------------------------------------------
// Node kernel: out = agg + eq_linear(agg, res) + eq_linear(nf, skip)
// ---------------------------------------------------------------------------
#if __has_builtin(__builtin_amdgcn_wmma_f32_16x16x4_f32)
#define HAS_WMMA_F32 1
#warning "CDNA5: node kernel using f32 WMMA 16x16x4 path"
#else
#define HAS_WMMA_F32 0
#warning "CDNA5: node kernel using VALU fallback (no f32 WMMA builtin)"
#endif

#if HAS_WMMA_F32
__device__ inline v8f wmma_f32(v2f a, v2f b, v8f c) {
    return __builtin_amdgcn_wmma_f32_16x16x4_f32(false, a, false, b, (short)0, c,
                                                 false, false);
}

__global__ __launch_bounds__(256) void node_kernel(
    const float* __restrict__ nf, const float* __restrict__ agg,
    const float* __restrict__ W0r, const float* __restrict__ W1r,
    const float* __restrict__ W0s, const float* __restrict__ W1s,
    float* __restrict__ out)
{
    __shared__ __attribute__((aligned(16))) float sA[16 * SROW];
    __shared__ __attribute__((aligned(16))) float sN[16 * SROW];
    const int tid = threadIdx.x;
    const int n0 = blockIdx.x * 16;
    for (int i = tid; i < 16 * DFULL; i += 256) {
        int row = i >> 9, col = i & 511;
        sA[row * SROW + col] = agg[(size_t)(n0 + row) * DFULL + col];
        sN[row * SROW + col] = nf [(size_t)(n0 + row) * DFULL + col];
    }
    __syncthreads();

    const int wave = tid >> 5, lane = tid & 31, m = lane & 15, hi = lane >> 4;
    for (int t = 0; t < 4; ++t) {
        const int tileId = wave * 4 + t;          // 32 output tiles per WG
        v8f c;
        if (tileId < 8) {                         // scalar channels, cols d0..d0+15
            const int d0 = tileId * 16;
#pragma unroll
            for (int r = 0; r < 8; ++r) c[r] = sA[(r + 8 * hi) * SROW + d0 + m];
            for (int ks = 0; ks < 32; ++ks) {     // agg @ res_W0
                const int k0 = ks * 4 + 2 * hi;   // A/B layout: VGPR v -> k0+v
                v2f a = *(const v2f*)&sA[m * SROW + k0];   // single ds_load_b64
                v2f b;
                b[0] = W0r[k0 * CCH + d0 + m] * F_LIN_SCALE;
                b[1] = W0r[(k0 + 1) * CCH + d0 + m] * F_LIN_SCALE;
                c = wmma_f32(a, b, c);
            }
            for (int ks = 0; ks < 32; ++ks) {     // nf @ skip_W0
                const int k0 = ks * 4 + 2 * hi;
                v2f a = *(const v2f*)&sN[m * SROW + k0];
                v2f b;
                b[0] = W0s[k0 * CCH + d0 + m] * F_LIN_SCALE;
                b[1] = W0s[(k0 + 1) * CCH + d0 + m] * F_LIN_SCALE;
                c = wmma_f32(a, b, c);
            }
#pragma unroll
            for (int r = 0; r < 8; ++r)
                out[(size_t)(n0 + r + 8 * hi) * DFULL + d0 + m] = c[r];
        } else {                                  // vector channels, component i3
            const int v  = tileId - 8;
            const int i3 = v >> 3;                // 0..2
            const int db = (v & 7) * 16;          // d block
            const int oc = CCH + (db + m) * 3 + i3;
#pragma unroll
            for (int r = 0; r < 8; ++r) c[r] = sA[(r + 8 * hi) * SROW + oc];
            for (int ks = 0; ks < 32; ++ks) {     // agg(x1,i3) @ res_W1
                const int k0 = ks * 4 + 2 * hi;
                v2f a, b;
                a[0] = sA[m * SROW + CCH + k0 * 3 + i3];
                a[1] = sA[m * SROW + CCH + (k0 + 1) * 3 + i3];
                b[0] = W1r[k0 * CCH + db + m] * F_LIN_SCALE;
                b[1] = W1r[(k0 + 1) * CCH + db + m] * F_LIN_SCALE;
                c = wmma_f32(a, b, c);
            }
            for (int ks = 0; ks < 32; ++ks) {     // nf(x1,i3) @ skip_W1
                const int k0 = ks * 4 + 2 * hi;
                v2f a, b;
                a[0] = sN[m * SROW + CCH + k0 * 3 + i3];
                a[1] = sN[m * SROW + CCH + (k0 + 1) * 3 + i3];
                b[0] = W1s[k0 * CCH + db + m] * F_LIN_SCALE;
                b[1] = W1s[(k0 + 1) * CCH + db + m] * F_LIN_SCALE;
                c = wmma_f32(a, b, c);
            }
#pragma unroll
            for (int r = 0; r < 8; ++r)
                out[(size_t)(n0 + r + 8 * hi) * DFULL + oc] = c[r];
        }
    }
}
#else
__global__ __launch_bounds__(256) void node_kernel(
    const float* __restrict__ nf, const float* __restrict__ agg,
    const float* __restrict__ W0r, const float* __restrict__ W1r,
    const float* __restrict__ W0s, const float* __restrict__ W1s,
    float* __restrict__ out)
{
    __shared__ float sA[16 * SROW];
    __shared__ float sN[16 * SROW];
    const int tid = threadIdx.x;
    const int n0 = blockIdx.x * 16;
    for (int i = tid; i < 16 * DFULL; i += 256) {
        int row = i >> 9, col = i & 511;
        sA[row * SROW + col] = agg[(size_t)(n0 + row) * DFULL + col];
        sN[row * SROW + col] = nf [(size_t)(n0 + row) * DFULL + col];
    }
    __syncthreads();
    for (int i = tid; i < 16 * DFULL; i += 256) {
        int row = i >> 9, col = i & 511;
        float s = 0.f, u = 0.f;
        if (col < CCH) {
            for (int cc = 0; cc < CCH; ++cc) {
                s += sA[row * SROW + cc] * W0r[cc * CCH + col];
                u += sN[row * SROW + cc] * W0s[cc * CCH + col];
            }
        } else {
            int cc2 = col - CCH, dd = cc2 / 3, ii = cc2 % 3;
            for (int cc = 0; cc < CCH; ++cc) {
                s += sA[row * SROW + CCH + cc * 3 + ii] * W1r[cc * CCH + dd];
                u += sN[row * SROW + CCH + cc * 3 + ii] * W1s[cc * CCH + dd];
            }
        }
        out[(size_t)(n0 + row) * DFULL + col] =
            sA[row * SROW + col] + F_LIN_SCALE * (s + u);
    }
}
#endif

// ---------------------------------------------------------------------------
extern "C" void kernel_launch(void* const* d_in, const int* in_sizes, int n_in,
                              void* d_out, int out_size, void* d_ws, size_t ws_size,
                              hipStream_t stream) {
    const float* nf    = (const float*)d_in[0];
    const float* ef    = (const float*)d_in[1];
    const float* attrs = (const float*)d_in[2];
    const float* emb   = (const float*)d_in[3];
    const int*   snd   = (const int*)  d_in[4];
    const int*   rcv   = (const int*)  d_in[5];
    const float* Wm1   = (const float*)d_in[6];
    const float* Wm2   = (const float*)d_in[7];
    const float* Wm3   = (const float*)d_in[8];
    const float* Wm4   = (const float*)d_in[9];
    const float* W0r   = (const float*)d_in[10];
    const float* W1r   = (const float*)d_in[11];
    const float* W0s   = (const float*)d_in[12];
    const float* W1s   = (const float*)d_in[13];
    float* out = (float*)d_out;
    float* agg = (float*)d_ws;                   // N x 512 f32 = 20.5 MB scratch

    const int aggN = NODES * DFULL;
    zero_kernel<<<(aggN + 255) / 256, 256, 0, stream>>>(agg, aggN);

    const int nTiles = EDGES / 16;               // 10000
    edge_kernel<<<nTiles / 8, 256, 0, stream>>>(nf, ef, attrs, emb, snd, rcv,
                                                Wm1, Wm2, Wm3, Wm4, agg);

    node_kernel<<<NODES / 16, 256, 0, stream>>>(nf, agg, W0r, W1r, W0s, W1s, out);
}
